// MultiHeadDSRA2_7344394076317
// MI455X (gfx1250) — matrix operand order repaired and checked by reading, not compiled, not run on hardware
//
#include <hip/hip_runtime.h>
#include <hip/hip_bf16.h>

#define BB 2
#define TT 2048
#define DD 1024
#define HH 8
#define DHH 128
#define KK 128
#define RR 8
#define WTOP 4

typedef __bf16 bf16_t;
typedef bf16_t bf16x16 __attribute__((ext_vector_type(16)));
typedef float f32x8 __attribute__((ext_vector_type(8)));
typedef unsigned short ushort_t;

#if defined(__has_builtin)
#if __has_builtin(__builtin_amdgcn_tensor_load_to_lds) && \
    __has_builtin(__builtin_amdgcn_s_wait_tensorcnt)
#define USE_TDM 1
#endif
#endif

#ifdef USE_TDM
typedef unsigned int u32x4 __attribute__((ext_vector_type(4)));
typedef int i32x4 __attribute__((ext_vector_type(4)));
typedef int i32x8 __attribute__((ext_vector_type(8)));

// Issue a 2D TDM tile load (bf16 data, no conversion) global -> LDS.
// D# per CDNA5 ISA 8.3/8.4: group0 = {count=1, lds_addr, global_addr, type=2};
// group1 packs data_size=2B + tensor dims/tile dims/stride across bit fields.
// 6-arg builtin (clang-23 / therock headers): (g0, g1, g2, g3, g4, cpol).
__device__ __forceinline__ void tdm_load_2d(unsigned lds_off, const void* gptr,
                                            int w1, int w2, int w3, int w4, int w5) {
  unsigned long long ga = (unsigned long long)(uintptr_t)gptr;
  u32x4 g0 = {1u,                                   // count=1, user mode
              lds_off,                              // D#.lds_addr (bytes)
              (unsigned)ga,                         // global_addr[31:0]
              ((unsigned)(ga >> 32) & 0x01FFFFFFu)  // global_addr[56:32]
                  | 0x80000000u};                   // type=2 ("image")
  i32x8 g1 = {0x10000,  // workgroup_mask=0, data_size=1 (2 bytes)
              w1,       // tensor_dim0[15:0] << 16
              w2,       // tensor_dim0[31:16] | tensor_dim1[15:0]<<16
              w3,       // tensor_dim1[31:16] | tile_dim0<<16
              w4,       // tile_dim1 | tile_dim2<<16
              w5,       // tensor_dim0_stride[31:0]
              0, 0};
  i32x4 z4 = {0, 0, 0, 0};          // 2D tensor: groups 2/3 unused
  i32x8 z8 = {0, 0, 0, 0, 0, 0, 0, 0};
  __builtin_amdgcn_tensor_load_to_lds(g0, g1, z4, z4, z8, 0);
}
#endif

union Frag16 { ushort_t us[16]; bf16x16 v; };

__device__ __forceinline__ ushort_t f2bf(float f) {
  union { float f; unsigned u; } c; c.f = f;
  unsigned u = c.u;
  unsigned r = u + 0x7FFFu + ((u >> 16) & 1u);  // round-to-nearest-even
  return (ushort_t)(r >> 16);
}

__device__ __forceinline__ f32x8 wmma_bf16(bf16x16 a, bf16x16 b, f32x8 c) {
  return __builtin_amdgcn_wmma_f32_16x16x32_bf16(false, a, false, b, (short)0, c,
                                                 false, false);
}

// ---- WMMA fragment loaders (ISA 7.12.2 layouts, wave32) ----
// A 16x32 bf16: lane holds row m=lane&15; half=lane>>4; v0..3 K=half*8+0..7, v4..7 K=16+half*8+0..7
__device__ __forceinline__ bf16x16 a_frag_lds(const ushort_t* t, int ldk, int row0,
                                              int koff, int lane) {
  const int m = row0 + (lane & 15);
  const int half = (lane >> 4) & 1;
  const ushort_t* p = t + m * ldk + koff;
  Frag16 f;
#pragma unroll
  for (int i = 0; i < 8; ++i) f.us[i] = p[half * 8 + i];
#pragma unroll
  for (int i = 0; i < 8; ++i) f.us[8 + i] = p[16 + half * 8 + i];
  return f.v;
}
__device__ __forceinline__ bf16x16 a_frag_g(const ushort_t* rowp, int koff, int lane) {
  const int half = (lane >> 4) & 1;
  const ushort_t* p = rowp + koff;
  Frag16 f;
#pragma unroll
  for (int i = 0; i < 8; ++i) f.us[i] = p[half * 8 + i];
#pragma unroll
  for (int i = 0; i < 8; ++i) f.us[8 + i] = p[16 + half * 8 + i];
  return f.v;
}
__device__ __forceinline__ bf16x16 a_frag_f32(const float* rowp, int koff, int lane,
                                              float s) {
  const int half = (lane >> 4) & 1;
  const float* p = rowp + koff;
  Frag16 f;
#pragma unroll
  for (int i = 0; i < 8; ++i) f.us[i] = f2bf(p[half * 8 + i] * s);
#pragma unroll
  for (int i = 0; i < 8; ++i) f.us[8 + i] = f2bf(p[16 + half * 8 + i] * s);
  return f.v;
}
// B 32x16 bf16 from N-major LDS tile [N][ldk]: lane col n=lane&15; v0..3 K=half*8+0..7, v4..7 K=16+half*8+0..7
__device__ __forceinline__ bf16x16 b_frag_lds(const ushort_t* t, int ldk, int col0,
                                              int koff, int lane) {
  const int n = col0 + (lane & 15);
  const int half = (lane >> 4) & 1;
  const ushort_t* p = t + n * ldk + koff;
  Frag16 f;
#pragma unroll
  for (int i = 0; i < 8; ++i) f.us[i] = p[half * 8 + i];
#pragma unroll
  for (int i = 0; i < 8; ++i) f.us[8 + i] = p[16 + half * 8 + i];
  return f.v;
}

// ---- wave32 reductions ----
__device__ __forceinline__ float wave_sum(float v) {
#pragma unroll
  for (int m = 16; m >= 1; m >>= 1) v += __shfl_xor(v, m, 32);
  return v;
}
__device__ __forceinline__ float wave_max(float v) {
#pragma unroll
  for (int m = 16; m >= 1; m >>= 1) v = fmaxf(v, __shfl_xor(v, m, 32));
  return v;
}
__device__ __forceinline__ float half_sum16(float v) {
#pragma unroll
  for (int m = 8; m >= 1; m >>= 1) v += __shfl_xor(v, m, 32);
  return v;
}
__device__ __forceinline__ float half_max16(float v) {
#pragma unroll
  for (int m = 8; m >= 1; m >>= 1) v = fmaxf(v, __shfl_xor(v, m, 32));
  return v;
}

// ======================= GEMM (bf16 WMMA, fp32 acc) =======================
// MODE 0: A=x f32 [4096x1024], B=w_qkv [1024x3072], scatter q/k/v head layout
// MODE 1: A=fused bf16 [4096x1024] (TDM-staged), B=w_out [1024x1024], write y
template <int MODE>
__global__ void __launch_bounds__(256) gemm_bf16_kernel(
    const void* __restrict__ Asrc, const float* __restrict__ Bsrc,
    float* __restrict__ qb, float* __restrict__ kb, float* __restrict__ vb,
    float* __restrict__ yout, int Kdim, int Ndim) {
  __shared__ ushort_t As[64 * 32];
  __shared__ ushort_t Bs[128 * 32];
  const int tid = threadIdx.x;
  const int lane = tid & 31;
  const int w = tid >> 5;
  const int wm = w & 3, wn = w >> 2;
  const int m0 = blockIdx.y * 64;
  const int n0 = blockIdx.x * 128;
  f32x8 acc[4] = {};
  const int arow = tid >> 2;
  const int acb = (tid & 3) * 8;
  for (int k0 = 0; k0 < Kdim; k0 += 32) {
    if (MODE == 0) {
      const float* s = (const float*)Asrc + (size_t)(m0 + arow) * Kdim + k0 + acb;
#pragma unroll
      for (int i = 0; i < 8; ++i) As[arow * 32 + acb + i] = f2bf(s[i]);
    } else {
#ifdef USE_TDM
      // DMA the bf16 A tile (64 rows x 32 cols, row stride Kdim) into LDS;
      // overlaps with the VALU f32->bf16 staging of the B tile below.
      if (w == 0)
        tdm_load_2d((unsigned)(uintptr_t)As,
                    (const ushort_t*)Asrc + (size_t)m0 * Kdim + k0,
                    (Kdim & 0xFFFF) << 16,                    // tensor_dim0 = Kdim
                    (Kdim >> 16) | (int)(0xFFFFu << 16),      // tensor_dim1 = 65535
                    32 << 16,                                 // tile_dim0 = 32
                    64,                                       // tile_dim1 = 64
                    Kdim);                                    // dim0 stride
#else
      const ushort_t* s = (const ushort_t*)Asrc + (size_t)(m0 + arow) * Kdim + k0 + acb;
#pragma unroll
      for (int i = 0; i < 8; ++i) As[arow * 32 + acb + i] = s[i];
#endif
    }
#pragma unroll
    for (int i = 0; i < 16; ++i) {
      int li = tid + 256 * i;
      int kk = li >> 7, n = li & 127;
      Bs[n * 32 + kk] = f2bf(Bsrc[(size_t)(k0 + kk) * Ndim + n0 + n]);
    }
    if (k0 + 32 < Kdim)  // speculative prefetch of next B k-slab -> global_prefetch_b8
      __builtin_prefetch(Bsrc + (size_t)(k0 + 32 + (tid >> 7)) * Ndim + n0 + (tid & 127), 0, 1);
#ifdef USE_TDM
    if (MODE == 1) __builtin_amdgcn_s_wait_tensorcnt(0);
#endif
    __syncthreads();
    bf16x16 a = a_frag_lds(As, 32, wm * 16, 0, lane);
#pragma unroll
    for (int j = 0; j < 4; ++j) {
      bf16x16 b = b_frag_lds(Bs, 32, wn * 64 + j * 16, 0, lane);
      acc[j] = wmma_bf16(a, b, acc[j]);
    }
    __syncthreads();
  }
  const int half = lane >> 4, nl = lane & 15;
#pragma unroll
  for (int j = 0; j < 4; ++j) {
#pragma unroll
    for (int r = 0; r < 8; ++r) {
      int row = m0 + wm * 16 + r + 8 * half;
      int col = n0 + wn * 64 + j * 16 + nl;
      float v = acc[j][r];
      if (MODE == 0) {
        int b = row >> 11, t = row & (TT - 1);
        int part = col >> 10, dg = col & 1023;
        int h = dg >> 7, dh = dg & 127;
        float* dst = (part == 0) ? qb : (part == 1) ? kb : vb;
        dst[(((size_t)b * HH + h) * TT + t) * DHH + dh] = v;
      } else {
        yout[(size_t)row * Ndim + col] = v;
      }
    }
  }
}

// ======================= slot logits: An[bh] (TxDH) @ skn[h]^T -> (TxK) ===
__global__ void __launch_bounds__(128) slot_logits_kernel(
    const ushort_t* __restrict__ An, const ushort_t* __restrict__ skn,
    float* __restrict__ outp, const float* __restrict__ log_tau, float addc) {
  __shared__ ushort_t Sk[KK * DHH];  // 32KB, N-major == [slot][dh]
  const int tid = threadIdx.x, lane = tid & 31, w = tid >> 5;
  const int bh = blockIdx.y, h = bh & (HH - 1);
  const int row0 = blockIdx.x * 64 + w * 16;
#ifdef USE_TDM
  // One TDM descriptor moves the whole 32KB bf16 skn[h] tile (16384 contiguous
  // elements) into LDS: tensor_dim0 = tile_dim0 = 16384, one row.
  if (w == 0)
    tdm_load_2d((unsigned)(uintptr_t)Sk, skn + (size_t)h * KK * DHH,
                0x4000 << 16,        // tensor_dim0 = 16384
                1 << 16,             // tensor_dim1 = 1
                0x4000 << 16,        // tile_dim0 = 16384
                1,                   // tile_dim1 = 1
                16384);              // dim0 stride
  __builtin_amdgcn_s_wait_tensorcnt(0);
#else
  for (int i = tid; i < KK * DHH; i += 128) Sk[i] = skn[(size_t)h * KK * DHH + i];
#endif
  __syncthreads();
  const ushort_t* arow = An + ((size_t)bh * TT + row0 + (lane & 15)) * DHH;
  f32x8 acc[8] = {};
#pragma unroll
  for (int ks = 0; ks < 4; ++ks) {
    bf16x16 a = a_frag_g(arow, ks * 32, lane);
#pragma unroll
    for (int j = 0; j < 8; ++j) {
      bf16x16 b = b_frag_lds(Sk, DHH, j * 16, ks * 32, lane);
      acc[j] = wmma_bf16(a, b, acc[j]);
    }
  }
  float scale = (log_tau != nullptr) ? expf(*log_tau) : 1.0f;
  const int half = lane >> 4, nl = lane & 15;
#pragma unroll
  for (int j = 0; j < 8; ++j)
#pragma unroll
    for (int r = 0; r < 8; ++r) {
      int row = row0 + r + 8 * half;
      outp[((size_t)bh * TT + row) * KK + j * 16 + nl] = acc[j][r] * scale + addc;
    }
}

// ======================= flash causal attention (WMMA, online softmax) =====
__global__ void __launch_bounds__(128) flash_attn_kernel(
    const float* __restrict__ q, const float* __restrict__ k,
    const float* __restrict__ v, float* __restrict__ out_local) {
  __shared__ ushort_t Ks[64 * DHH];   // [kp][dh]    16KB
  __shared__ ushort_t Vs[DHH * 64];   // [dh][kp]    16KB (transposed)
  __shared__ ushort_t Ps[4 * 16 * 64];// per-wave P   8KB
  const int tid = threadIdx.x, lane = tid & 31, w = tid >> 5;
  const int qt = blockIdx.x, bh = blockIdx.y;
  const size_t base = (size_t)bh * TT * DHH;
  const int m0 = qt * 64 + w * 16;
  const int half = lane >> 4, nl = lane & 15;
  const float sc = 0.08838834764831845f;  // DH^-0.5, folded into Q
  bf16x16 aq[4];
  {
    const float* qrow = q + base + (size_t)(m0 + nl) * DHH;
#pragma unroll
    for (int ks = 0; ks < 4; ++ks) aq[ks] = a_frag_f32(qrow, ks * 32, lane, sc);
  }
  f32x8 of[8] = {};
  float mrow[8], lrow[8];
#pragma unroll
  for (int r = 0; r < 8; ++r) { mrow[r] = -1e30f; lrow[r] = 0.0f; }
  for (int kt = 0; kt <= qt; ++kt) {
    const int n0k = kt * 64;
    for (int i = 0; i < 64; ++i) {
      int li = tid + 128 * i;
      int rr = li >> 7, cc = li & 127;
      Ks[rr * DHH + cc] = f2bf(k[base + (size_t)(n0k + rr) * DHH + cc]);
      Vs[cc * 64 + rr] = f2bf(v[base + (size_t)(n0k + rr) * DHH + cc]);
    }
    __syncthreads();
    f32x8 s[4] = {};
#pragma unroll
    for (int ks = 0; ks < 4; ++ks) {
#pragma unroll
      for (int j = 0; j < 4; ++j) {
        bf16x16 b = b_frag_lds(Ks, DHH, j * 16, ks * 32, lane);
        s[j] = wmma_bf16(aq[ks], b, s[j]);
      }
    }
    if (kt == qt) {  // causal mask on diagonal tile
#pragma unroll
      for (int j = 0; j < 4; ++j)
#pragma unroll
        for (int r = 0; r < 8; ++r) {
          int rowg = m0 + r + 8 * half;
          int colg = n0k + j * 16 + nl;
          if (colg > rowg) s[j][r] = -1e30f;
        }
    }
#pragma unroll
    for (int r = 0; r < 8; ++r) {
      float lm = s[0][r];
#pragma unroll
      for (int j = 1; j < 4; ++j) lm = fmaxf(lm, s[j][r]);
      lm = half_max16(lm);
      float mnew = fmaxf(mrow[r], lm);
      float corr = expf(mrow[r] - mnew);
      mrow[r] = mnew;
      float rs = 0.0f;
#pragma unroll
      for (int j = 0; j < 4; ++j) {
        float p = expf(s[j][r] - mnew);
        s[j][r] = p;
        rs += p;
      }
      rs = half_sum16(rs);
      lrow[r] = lrow[r] * corr + rs;
#pragma unroll
      for (int f = 0; f < 8; ++f) of[f][r] *= corr;
    }
    // re-stage P into A-fragment layout via per-wave LDS (same-wave dep; no barrier)
#pragma unroll
    for (int j = 0; j < 4; ++j)
#pragma unroll
      for (int r = 0; r < 8; ++r)
        Ps[w * 1024 + (r + 8 * half) * 64 + j * 16 + nl] = f2bf(s[j][r]);
#pragma unroll
    for (int ks2 = 0; ks2 < 2; ++ks2) {
      bf16x16 pa = a_frag_lds(Ps + w * 1024, 64, 0, ks2 * 32, lane);
#pragma unroll
      for (int f = 0; f < 8; ++f) {
        bf16x16 vb = b_frag_lds(Vs, 64, f * 16, ks2 * 32, lane);
        of[f] = wmma_bf16(pa, vb, of[f]);
      }
    }
    __syncthreads();
  }
#pragma unroll
  for (int f = 0; f < 8; ++f)
#pragma unroll
    for (int r = 0; r < 8; ++r) {
      int rowg = m0 + r + 8 * half;
      out_local[base + (size_t)rowg * DHH + f * 16 + nl] = of[f][r] / lrow[r];
    }
}

// ======================= normalization kernels =======================
__global__ void __launch_bounds__(256) norm_slots_kernel(
    const float* __restrict__ slot_k_init, ushort_t* __restrict__ skn_bf,
    float* __restrict__ slotk_n) {
  const int lane = threadIdx.x & 31;
  const int row = blockIdx.x * 8 + (threadIdx.x >> 5);  // [0, H*K)
  const size_t off = (size_t)row * DHH + lane * 4;
  float x[4]; float ss = 0.0f;
#pragma unroll
  for (int i = 0; i < 4; ++i) { x[i] = slot_k_init[off + i]; ss += x[i] * x[i]; }
  ss = wave_sum(ss);
  float inv = 1.0f / fmaxf(sqrtf(ss), 1e-12f);
#pragma unroll
  for (int i = 0; i < 4; ++i) {
    float nv = x[i] * inv;
    skn_bf[off + i] = f2bf(nv);
    slotk_n[off + i] = nv;
  }
}

__global__ void __launch_bounds__(256) norm_qk_kernel(
    const float* __restrict__ q, const float* __restrict__ k,
    ushort_t* __restrict__ qn, ushort_t* __restrict__ kn) {
  const int lane = threadIdx.x & 31;
  const int row = blockIdx.x * 8 + (threadIdx.x >> 5);  // [0, B*H*T)
  const size_t off = (size_t)row * DHH + lane * 4;
  float x[4]; float ss = 0.0f;
#pragma unroll
  for (int i = 0; i < 4; ++i) { x[i] = q[off + i]; ss += x[i] * x[i]; }
  ss = wave_sum(ss);
  float inv = 1.0f / fmaxf(sqrtf(ss), 1e-12f);
#pragma unroll
  for (int i = 0; i < 4; ++i) qn[off + i] = f2bf(x[i] * inv);
  ss = 0.0f;
#pragma unroll
  for (int i = 0; i < 4; ++i) { x[i] = k[off + i]; ss += x[i] * x[i]; }
  ss = wave_sum(ss);
  inv = 1.0f / fmaxf(sqrtf(ss), 1e-12f);
#pragma unroll
  for (int i = 0; i < 4; ++i) kn[off + i] = f2bf(x[i] * inv);
}

// ======================= read branch + fuse (wave per token) =============
__global__ void __launch_bounds__(256) read_fuse_kernel(
    const float* __restrict__ rlog, const float* __restrict__ q,
    const float* __restrict__ slot_v, const float* __restrict__ slotk_n,
    const float* __restrict__ out_local, const float* __restrict__ w_fuse,
    const float* __restrict__ b_fuse, ushort_t* __restrict__ fusedA) {
  const int lane = threadIdx.x & 31;
  const int gw = blockIdx.x * 8 + (threadIdx.x >> 5);  // [0, B*H*T)
  const int bh = gw >> 11;
  const int t = gw & (TT - 1);
  const int h = bh & (HH - 1);
  const int b = bh >> 3;
  float la[4];
  const size_t rowoff = (size_t)gw * KK;
#pragma unroll
  for (int i = 0; i < 4; ++i) la[i] = rlog[rowoff + lane * 4 + i];
  float topv[RR]; int topi[RR];
#pragma unroll
  for (int r = 0; r < RR; ++r) {
    float bv = -1e30f; int bi = 1 << 30;
#pragma unroll
    for (int i = 0; i < 4; ++i) {
      int idx = lane * 4 + i;
      if (la[i] > bv || (la[i] == bv && idx < bi)) { bv = la[i]; bi = idx; }
    }
#pragma unroll
    for (int m = 16; m >= 1; m >>= 1) {
      float ov = __shfl_xor(bv, m, 32);
      int oi = __shfl_xor(bi, m, 32);
      if (ov > bv || (ov == bv && oi < bi)) { bv = ov; bi = oi; }
    }
    topv[r] = bv; topi[r] = bi;
#pragma unroll
    for (int i = 0; i < 4; ++i)
      if (bi == lane * 4 + i) la[i] = -1e30f;
  }
  float pr[RR]; float psum = 0.0f;
#pragma unroll
  for (int r = 0; r < RR; ++r) { pr[r] = expf(topv[r] - topv[0]); psum += pr[r]; }
  float pinv = 1.0f / psum;
  float sv[RR][4], sk[RR][4];
#pragma unroll
  for (int r = 0; r < RR; ++r) {
    const float* vp = slot_v + ((size_t)h * KK + topi[r]) * DHH + lane * 4;
    const float* kp = slotk_n + ((size_t)h * KK + topi[r]) * DHH + lane * 4;
#pragma unroll
    for (int i = 0; i < 4; ++i) { sv[r][i] = vp[i]; sk[r][i] = kp[i]; }
  }
  float qv[4];
  const size_t qoff = (size_t)gw * DHH + lane * 4;
#pragma unroll
  for (int i = 0; i < 4; ++i) qv[i] = q[qoff + i];
  float rw[RR]; float rwsum = 0.0f;
#pragma unroll
  for (int r = 0; r < RR; ++r) {
    float part = 0.0f;
#pragma unroll
    for (int i = 0; i < 4; ++i) part += qv[i] * sk[r][i];
    part = wave_sum(part) * 0.08838834764831845f;
    float sg = 1.0f / (1.0f + expf(-2.0f * part));
    rw[r] = sg; rwsum += sg;
  }
  float rinv = 1.0f / fmaxf(rwsum, 1e-6f);
  float oread[4] = {0, 0, 0, 0}, oret[4] = {0, 0, 0, 0};
#pragma unroll
  for (int r = 0; r < RR; ++r) {
    float p = pr[r] * pinv;
    float w2 = rw[r] * rinv;
#pragma unroll
    for (int i = 0; i < 4; ++i) { oread[i] += p * sv[r][i]; oret[i] += w2 * sv[r][i]; }
  }
  float g[3];
#pragma unroll
  for (int j = 0; j < 3; ++j) {
    float part = 0.0f;
#pragma unroll
    for (int i = 0; i < 4; ++i) part += qv[i] * w_fuse[(lane * 4 + i) * 3 + j];
    g[j] = wave_sum(part) + b_fuse[j];
  }
  float gm = fmaxf(g[0], fmaxf(g[1], g[2]));
  float e0 = expf(g[0] - gm), e1 = expf(g[1] - gm), e2 = expf(g[2] - gm);
  float gs = 1.0f / (e0 + e1 + e2);
  e0 *= gs; e1 *= gs; e2 *= gs;
  ushort_t* dst = fusedA + ((size_t)b * TT + t) * DD + h * DHH + lane * 4;
#pragma unroll
  for (int i = 0; i < 4; ++i) {
    float fu = e0 * out_local[qoff + i] + e1 * oread[i] + e2 * oret[i];
    dst[i] = f2bf(fu);
  }
}

// ======================= write branch (wave per token, atomic scatter) ====
__global__ void __launch_bounds__(256) write_topk_kernel(
    const float* __restrict__ sim, const float* __restrict__ kbuf,
    const float* __restrict__ vbuf, const float* __restrict__ w_twg,
    const float* __restrict__ b_twg, const float* __restrict__ log_tau_w,
    float* __restrict__ val_acc, float* __restrict__ mass) {
  const int lane = threadIdx.x & 31;
  const int gw = blockIdx.x * 8 + (threadIdx.x >> 5);
  const int bh = gw >> 11;
  float sa[4];
  const size_t rowoff = (size_t)gw * KK;
#pragma unroll
  for (int i = 0; i < 4; ++i) sa[i] = sim[rowoff + lane * 4 + i];
  float mx = fmaxf(fmaxf(sa[0], sa[1]), fmaxf(sa[2], sa[3]));
  mx = wave_max(mx);
  float novelty = fminf(fmaxf(1.0f - mx, 0.0f), 1.0f);
  float tauw = expf(*log_tau_w);
  float wl[4];
#pragma unroll
  for (int i = 0; i < 4; ++i) wl[i] = sa[i] * tauw;
  float topv[WTOP]; int topi[WTOP];
#pragma unroll
  for (int r = 0; r < WTOP; ++r) {
    float bv = -1e30f; int bi = 1 << 30;
#pragma unroll
    for (int i = 0; i < 4; ++i) {
      int idx = lane * 4 + i;
      if (wl[i] > bv || (wl[i] == bv && idx < bi)) { bv = wl[i]; bi = idx; }
    }
#pragma unroll
    for (int m = 16; m >= 1; m >>= 1) {
      float ov = __shfl_xor(bv, m, 32);
      int oi = __shfl_xor(bi, m, 32);
      if (ov > bv || (ov == bv && oi < bi)) { bv = ov; bi = oi; }
    }
    topv[r] = bv; topi[r] = bi;
#pragma unroll
    for (int i = 0; i < 4; ++i)
      if (bi == lane * 4 + i) wl[i] = -1e30f;
  }
  float pe[WTOP]; float ps = 0.0f;
#pragma unroll
  for (int r = 0; r < WTOP; ++r) { pe[r] = expf(topv[r] - topv[0]); ps += pe[r]; }
  float pinv = 1.0f / ps;
  float kv[4], vv[4];
  const size_t koff = (size_t)gw * DHH + lane * 4;
#pragma unroll
  for (int i = 0; i < 4; ++i) { kv[i] = kbuf[koff + i]; vv[i] = vbuf[koff + i]; }
  float part = 0.0f;
#pragma unroll
  for (int i = 0; i < 4; ++i) part += kv[i] * w_twg[lane * 4 + i];
  part = wave_sum(part);
  float gate = 1.0f / (1.0f + expf(-(part + b_twg[0])));
#pragma unroll
  for (int r = 0; r < WTOP; ++r) {
    float ww = pe[r] * pinv * gate * novelty;
    float* dst = val_acc + ((size_t)bh * KK + topi[r]) * DHH + lane * 4;
#pragma unroll
    for (int i = 0; i < 4; ++i) atomicAdd(dst + i, ww * vv[i]);
    if (lane == 0) atomicAdd(mass + (size_t)bh * KK + topi[r], ww);
  }
}

// ======================= EMA finalize =======================
__global__ void __launch_bounds__(256) finalize_kernel(
    const float* __restrict__ val_acc, const float* __restrict__ mass,
    const float* __restrict__ slot_v, float* __restrict__ outv) {
  int idx = blockIdx.x * 256 + threadIdx.x;  // [0, B*H*K*DH)
  int bhk = idx >> 7;
  float m = mass[bhk];
  float upd = val_acc[idx] / (m + 1e-6f);
  float alpha = fminf(fmaxf(0.25f * m, 0.0f), 0.5f);
  int hkd = (bhk & (HH * KK - 1)) * DHH + (idx & 127);
  outv[idx] = (1.0f - alpha) * slot_v[hkd] + alpha * upd;
}

extern "C" void kernel_launch(void* const* d_in, const int* in_sizes, int n_in,
                              void* d_out, int out_size, void* d_ws, size_t ws_size,
                              hipStream_t stream) {
  (void)in_sizes; (void)n_in; (void)out_size; (void)ws_size;
  const float* x            = (const float*)d_in[0];
  const float* w_qkv        = (const float*)d_in[1];
  const float* w_out        = (const float*)d_in[2];
  const float* slot_k_init  = (const float*)d_in[3];
  const float* slot_v_init  = (const float*)d_in[4];
  const float* w_twg        = (const float*)d_in[5];
  const float* b_twg        = (const float*)d_in[6];
  const float* w_fuse       = (const float*)d_in[7];
  const float* b_fuse       = (const float*)d_in[8];
  const float* log_tau_read = (const float*)d_in[9];
  const float* log_tau_write= (const float*)d_in[10];

  char* ws = (char*)d_ws;
  size_t off = 0;
  auto alloc = [&](size_t bytes) -> void* {
    void* p = ws + off;
    off += (bytes + 255) & ~(size_t)255;
    return p;
  };
  float*    qb       = (float*)alloc((size_t)BB * HH * TT * DHH * 4);   // 16MB
  float*    kb       = (float*)alloc((size_t)BB * HH * TT * DHH * 4);
  float*    vb       = (float*)alloc((size_t)BB * HH * TT * DHH * 4);
  ushort_t* qn       = (ushort_t*)alloc((size_t)BB * HH * TT * DHH * 2);
  ushort_t* kn       = (ushort_t*)alloc((size_t)BB * HH * TT * DHH * 2);
  ushort_t* sknb     = (ushort_t*)alloc((size_t)HH * KK * DHH * 2);
  float*    slotk_n  = (float*)alloc((size_t)HH * KK * DHH * 4);
  float*    rlog     = (float*)alloc((size_t)BB * HH * TT * KK * 4);
  float*    simb     = (float*)alloc((size_t)BB * HH * TT * KK * 4);
  float*    out_loc  = (float*)alloc((size_t)BB * HH * TT * DHH * 4);
  ushort_t* fusedA   = (ushort_t*)alloc((size_t)BB * TT * DD * 2);
  float*    val_acc  = (float*)alloc((size_t)BB * HH * KK * DHH * 4);   // 1MB
  float*    mass     = (float*)alloc((size_t)BB * HH * KK * 4);         // 8KB (contiguous after val_acc)

  float* y = (float*)d_out;
  float* new_slot_v = (float*)d_out + (size_t)BB * TT * DD;

  // zero the scatter accumulators (val_acc + mass are contiguous, 256B aligned)
  (void)hipMemsetAsync(val_acc, 0,
                       (size_t)BB * HH * KK * DHH * 4 + (size_t)BB * HH * KK * 4, stream);

  norm_slots_kernel<<<HH * KK / 8, 256, 0, stream>>>(slot_k_init, sknb, slotk_n);
  gemm_bf16_kernel<0><<<dim3(3 * DD / 128, BB * TT / 64), 256, 0, stream>>>(
      x, w_qkv, qb, kb, vb, nullptr, DD, 3 * DD);
  norm_qk_kernel<<<BB * HH * TT / 8, 256, 0, stream>>>(qb, kb, qn, kn);
  slot_logits_kernel<<<dim3(TT / 64, BB * HH), 128, 0, stream>>>(
      qn, sknb, rlog, log_tau_read, 0.25f);  // CONF_RB*0.5 (fresh conf), age=0
  slot_logits_kernel<<<dim3(TT / 64, BB * HH), 128, 0, stream>>>(
      kn, sknb, simb, nullptr, 0.0f);        // raw sim for write path
  flash_attn_kernel<<<dim3(TT / 64, BB * HH), 128, 0, stream>>>(qb, kb, vb, out_loc);
  read_fuse_kernel<<<BB * HH * TT / 8, 256, 0, stream>>>(
      rlog, qb, slot_v_init, slotk_n, out_loc, w_fuse, b_fuse, fusedA);
  gemm_bf16_kernel<1><<<dim3(DD / 128, BB * TT / 64), 256, 0, stream>>>(
      fusedA, w_out, nullptr, nullptr, nullptr, y, DD, DD);
  write_topk_kernel<<<BB * HH * TT / 8, 256, 0, stream>>>(
      simb, kb, vb, w_twg, b_twg, log_tau_write, val_acc, mass);
  finalize_kernel<<<BB * HH * KK * DHH / 256, 256, 0, stream>>>(
      val_acc, mass, slot_v_init, new_slot_v);
}
